// W_Attention_79001628442779
// MI455X (gfx1250) — matrix operand order repaired and checked by reading, not compiled, not run on hardware
//
#include <hip/hip_runtime.h>
#include <hip/hip_bf16.h>

// Swin window attention, fused per-window kernel for gfx1250 (MI455X).
// 1 block = 1 window (8192 blocks), 128 threads = 4 wave32, wave h = head h.
// LDS ~107 KB -> 3 blocks / WGP for latency hiding (memory-bound problem).

#define WSZ   8
#define NTOK  64
#define CDIM  128
#define DH    32
#define HEADS 4

#define QSTR  40    // q/k/v LDS row stride (halfs): 20 dwords -> conflict-free frag reads
#define PSTR  72    // P-tile LDS row stride (halfs): 36 dwords
#define XSTR  136   // X / W LDS row stride (halfs): 68 dwords

typedef __attribute__((ext_vector_type(16))) _Float16 v16h;
typedef __attribute__((ext_vector_type(8)))  float    v8f;

union FragU {
    unsigned int u[8];
    _Float16     e[16];
    v16h         h;
};

// ds_swizzle xor-shuffle (group-of-32 mode): imm = (xor<<10) | (or<<5) | and(0x1f)
template <int IMM>
__device__ __forceinline__ float swz_xor(float x) {
    return __builtin_bit_cast(float,
        __builtin_amdgcn_ds_swizzle(__builtin_bit_cast(int, x), IMM));
}

__launch_bounds__(128)
__global__ void swin_window_attn_kernel(const float* __restrict__ q,
                                        const float* __restrict__ k,
                                        const float* __restrict__ v,
                                        const float* __restrict__ pw,
                                        const float* __restrict__ pb,
                                        const float* __restrict__ bt,
                                        float* __restrict__ out)
{
    __shared__ _Float16 qlds[HEADS][NTOK][QSTR];                 // 20480 B
    __shared__ __align__(16) _Float16 kxbuf[HEADS * NTOK * QSTR]; // 20480 B (K staging, then X)
    __shared__ _Float16 vlds[HEADS][NTOK][QSTR];                 // 20480 B
    __shared__ _Float16 ptile[HEADS][16][PSTR];                  //  9216 B (one 16-row P tile)
    __shared__ _Float16 wlds[CDIM][XSTR];                        // 34816 B
    __shared__ _Float16 btab[225 * HEADS];                       //  1800 B

    _Float16 (*klds)[NTOK][QSTR] = (_Float16 (*)[NTOK][QSTR])kxbuf; // [HEADS][NTOK][QSTR]
    _Float16 (*xlds)[XSTR]       = (_Float16 (*)[XSTR])kxbuf;       // [NTOK][XSTR] (8704 halfs)

    const int tid  = threadIdx.x;
    const int lane = tid & 31;
    const int h    = tid >> 5;          // wave index == head
    const int l16  = lane & 15;
    const int hi   = lane >> 4;
    const int win  = blockIdx.x;

    // ------------------------------------------------------------------
    // Stage 0: cooperative staging, f32 global -> f16 LDS (coalesced b128)
    // ------------------------------------------------------------------
    const size_t wbase = (size_t)win * NTOK * CDIM;
    const float4* q4 = (const float4*)(q + wbase);
    const float4* k4 = (const float4*)(k + wbase);
    const float4* v4 = (const float4*)(v + wbase);
    #pragma unroll
    for (int it = 0; it < 16; ++it) {               // 2048 float4 / 128 threads
        int idx  = it * 128 + tid;
        int tok  = idx >> 5;                        // 32 float4 per 128-ch row
        int c4   = idx & 31;
        int head = c4 >> 3;
        int colh = (c4 & 7) * 4;
        float4 fq = q4[idx], fk = k4[idx], fv = v4[idx];
        _Float16* dq = &qlds[head][tok][colh];
        dq[0] = (_Float16)fq.x; dq[1] = (_Float16)fq.y;
        dq[2] = (_Float16)fq.z; dq[3] = (_Float16)fq.w;
        _Float16* dk = &klds[head][tok][colh];
        dk[0] = (_Float16)fk.x; dk[1] = (_Float16)fk.y;
        dk[2] = (_Float16)fk.z; dk[3] = (_Float16)fk.w;
        _Float16* dv = &vlds[head][tok][colh];
        dv[0] = (_Float16)fv.x; dv[1] = (_Float16)fv.y;
        dv[2] = (_Float16)fv.z; dv[3] = (_Float16)fv.w;
    }
    const float4* w4 = (const float4*)pw;
    #pragma unroll
    for (int it = 0; it < 32; ++it) {               // 4096 float4 / 128 threads
        int idx = it * 128 + tid;
        int row = idx >> 5;
        int col = (idx & 31) * 4;
        float4 f = w4[idx];
        _Float16* dw = &wlds[row][col];
        dw[0] = (_Float16)f.x; dw[1] = (_Float16)f.y;
        dw[2] = (_Float16)f.z; dw[3] = (_Float16)f.w;
    }
    for (int i = tid; i < 225 * HEADS; i += 128) btab[i] = (_Float16)bt[i];
    __syncthreads();

    // ------------------------------------------------------------------
    // Load K / V fragments into registers, then free the K buffer for X.
    // ------------------------------------------------------------------
    FragU bk[4];                                    // B fragments of K^T
    #pragma unroll
    for (int tj = 0; tj < 4; ++tj) {
        const unsigned int* src = (const unsigned int*)&klds[h][tj * 16 + l16][0];
        #pragma unroll
        for (int r = 0; r < 8; ++r) bk[tj].u[r] = src[r + 8 * hi];
    }
    FragU bv[2][2];                                 // B fragments of V
    #pragma unroll
    for (int tk = 0; tk < 2; ++tk)
        #pragma unroll
        for (int tn = 0; tn < 2; ++tn)
            #pragma unroll
            for (int r = 0; r < 8; ++r) {
                int k0 = tk * 32 + 2 * r + 16 * hi;
                bv[tk][tn].e[2 * r]     = vlds[h][k0    ][tn * 16 + l16];
                bv[tk][tn].e[2 * r + 1] = vlds[h][k0 + 1][tn * 16 + l16];
            }
    __syncthreads();   // all waves done reading klds -> safe to reuse as xlds

    // ------------------------------------------------------------------
    // Stage 1+2 fused per 16-row tile:
    //   S = (Q K^T)*scale -> softmax (registers) -> +bias -> P tile (f16 LDS)
    //   X rows = P tile @ V  (K = 64 -> 2-WMMA chain)
    // ------------------------------------------------------------------
    const float scale = 0.17677669529663687f;       // 1/sqrt(32)

    #pragma unroll
    for (int ti = 0; ti < 4; ++ti) {
        FragU aq;                                   // A fragment of Q
        {
            const unsigned int* src = (const unsigned int*)&qlds[h][ti * 16 + l16][0];
            #pragma unroll
            for (int r = 0; r < 8; ++r) {
                int d = r + 8 * hi;
                aq.u[r] = src[d];
            }
        }
        v8f c[4];
        #pragma unroll
        for (int tj = 0; tj < 4; ++tj) {
            v8f z = {};
            c[tj] = __builtin_amdgcn_wmma_f32_16x16x32_f16(
                false, aq.h, false, bk[tj].h, (short)0, z, false, false);
        }
        // softmax per row: row i = ti*16 + r + 8*hi; cols across {tj} x 16 lanes
        #pragma unroll
        for (int r = 0; r < 8; ++r) {
            float s0 = c[0][r] * scale, s1 = c[1][r] * scale;
            float s2 = c[2][r] * scale, s3 = c[3][r] * scale;
            float m = fmaxf(fmaxf(s0, s1), fmaxf(s2, s3));
            m = fmaxf(m, swz_xor<0x041F>(m));       // xor 1
            m = fmaxf(m, swz_xor<0x081F>(m));       // xor 2
            m = fmaxf(m, swz_xor<0x101F>(m));       // xor 4
            m = fmaxf(m, swz_xor<0x201F>(m));       // xor 8
            float e0 = __expf(s0 - m), e1 = __expf(s1 - m);
            float e2 = __expf(s2 - m), e3 = __expf(s3 - m);
            float sum = e0 + e1 + e2 + e3;
            sum += swz_xor<0x041F>(sum);
            sum += swz_xor<0x081F>(sum);
            sum += swz_xor<0x101F>(sum);
            sum += swz_xor<0x201F>(sum);
            float inv = 1.0f / sum;
            int i  = ti * 16 + r + 8 * hi;          // global row
            int rr = r + 8 * hi;                    // row within 16-row tile
            int iy = i >> 3, ix = i & 7;
            float e[4] = {e0, e1, e2, e3};
            #pragma unroll
            for (int tj = 0; tj < 4; ++tj) {
                int j  = tj * 16 + l16;
                int jy = j >> 3, jx = j & 7;
                int bidx = ((iy - jy + 7) * 15 + (ix - jx + 7)) * HEADS + h;
                float p = e[tj] * inv + (float)btab[bidx];  // bias AFTER softmax
                ptile[h][rr][j] = (_Float16)p;
            }
        }
        // P tile (16x64) @ V (64x32) -> X rows [ti*16, ti*16+16)
        FragU ap[2];
        const unsigned int* src = (const unsigned int*)&ptile[h][l16][0];
        #pragma unroll
        for (int tk = 0; tk < 2; ++tk)
            #pragma unroll
            for (int r = 0; r < 8; ++r) {
                int d = r + 8 * hi;
                ap[tk].u[r] = src[tk * 16 + d];
            }
        #pragma unroll
        for (int tn = 0; tn < 2; ++tn) {
            v8f acc = {};
            acc = __builtin_amdgcn_wmma_f32_16x16x32_f16(
                false, ap[0].h, false, bv[0][tn].h, (short)0, acc, false, false);
            acc = __builtin_amdgcn_wmma_f32_16x16x32_f16(
                false, ap[1].h, false, bv[1][tn].h, (short)0, acc, false, false);
            #pragma unroll
            for (int r = 0; r < 8; ++r)
                xlds[ti * 16 + r + 8 * hi][h * 32 + tn * 16 + l16] = (_Float16)acc[r];
        }
    }
    __syncthreads();

    // ------------------------------------------------------------------
    // Stage 3: out = X @ proj_w^T + proj_b, window-reversed store
    // wave h owns output channels [h*32, h*32+32)
    // ------------------------------------------------------------------
    FragU bw[4][2];                                 // B fragments of W^T
    #pragma unroll
    for (int tn = 0; tn < 2; ++tn) {
        const unsigned int* src = (const unsigned int*)&wlds[h * 32 + tn * 16 + l16][0];
        #pragma unroll
        for (int tk = 0; tk < 4; ++tk)
            #pragma unroll
            for (int r = 0; r < 8; ++r)
                bw[tk][tn].u[r] = src[tk * 16 + r + 8 * hi];
    }
    float pbias[2];
    pbias[0] = pb[h * 32 + l16];
    pbias[1] = pb[h * 32 + 16 + l16];

    const int wb = win >> 10;                       // batch
    const int hw = (win >> 5) & 31;                 // window row
    const int ww = win & 31;                        // window col

    #pragma unroll
    for (int ti = 0; ti < 4; ++ti) {
        FragU ax[4];                                // A fragments of X
        const unsigned int* src = (const unsigned int*)&xlds[ti * 16 + l16][0];
        #pragma unroll
        for (int tk = 0; tk < 4; ++tk)
            #pragma unroll
            for (int r = 0; r < 8; ++r) {
                int d = r + 8 * hi;
                ax[tk].u[r] = src[tk * 16 + d];
            }
        #pragma unroll
        for (int tn = 0; tn < 2; ++tn) {
            v8f acc = {};
            #pragma unroll
            for (int tk = 0; tk < 4; ++tk)
                acc = __builtin_amdgcn_wmma_f32_16x16x32_f16(
                    false, ax[tk].h, false, bw[tk][tn].h, (short)0, acc, false, false);
            int cc = h * 32 + tn * 16 + l16;
            #pragma unroll
            for (int r = 0; r < 8; ++r) {
                int m  = ti * 16 + r + 8 * hi;      // token in window
                int wy = m >> 3, wx = m & 7;
                size_t off = ((((size_t)wb * 256 + hw * 8 + wy) * 256) + ww * 8 + wx)
                             * (size_t)CDIM + cc;
                out[off] = acc[r] + pbias[tn];
            }
        }
    }
}

extern "C" void kernel_launch(void* const* d_in, const int* in_sizes, int n_in,
                              void* d_out, int out_size, void* d_ws, size_t ws_size,
                              hipStream_t stream) {
    (void)in_sizes; (void)n_in; (void)d_ws; (void)ws_size; (void)out_size;
    // setup_inputs order: x, q, k, v, proj_w, proj_b, rel_bias_table
    const float* q  = (const float*)d_in[1];
    const float* k  = (const float*)d_in[2];
    const float* v  = (const float*)d_in[3];
    const float* pw = (const float*)d_in[4];
    const float* pb = (const float*)d_in[5];
    const float* bt = (const float*)d_in[6];
    float* out = (float*)d_out;

    dim3 grid(8192);   // 8 * 32 * 32 windows
    dim3 block(128);   // 4 waves (one per head)
    hipLaunchKernelGGL(swin_window_attn_kernel, grid, block, 0, stream,
                       q, k, v, pw, pb, bt, out);
}
